// CausalSelfAttention_91087666414094
// MI455X (gfx1250) — matrix operand order repaired and checked by reading, not compiled, not run on hardware
//
#include <hip/hip_runtime.h>

// ---------------------------------------------------------------------------
// CDNA5 (gfx1250, wave32) causal self-attention:
//   QKV proj (bf16 WMMA GEMM, async-LDS pipelined) -> flash attention (WMMA,
//   64-key blocks) -> out proj. Global->LDS staging via
//   GLOBAL_LOAD_ASYNC_TO_LDS_B128; row reductions via DS_SWIZZLE_B32.
// ---------------------------------------------------------------------------

typedef __attribute__((ext_vector_type(16))) __bf16 v16bf;
typedef __attribute__((ext_vector_type(8)))  __bf16 v8bf;
typedef __attribute__((ext_vector_type(8)))  float  v8f;

union V16U { v16bf v; v8bf h[2]; };

constexpr int Bn  = 2;
constexpr int Tn  = 2048;
constexpr int Dn  = 1024;
constexpr int Hn  = 16;
constexpr int HDn = 64;
constexpr int Mrows = Bn * Tn;   // 4096 token rows

__device__ __forceinline__ v8f wmma_bf16(v16bf a, v16bf b, v8f c) {
  // D = A(16x32 bf16) * B(32x16 bf16) + C(16x16 f32)
  return __builtin_amdgcn_wmma_f32_16x16x32_bf16(
      false, a, false, b, (short)0, c, false, false);
}

// Async global -> LDS 16-byte copy (per lane). ASYNCcnt-tracked.
__device__ __forceinline__ void async_copy_b128(const void* lds_dst, const void* gsrc) {
  const unsigned int       l = (unsigned int)(unsigned long long)lds_dst; // addr[31:0]
  const unsigned long long g = (unsigned long long)gsrc;
  asm volatile("global_load_async_to_lds_b128 %0, %1, off"
               :: "v"(l), "v"(g) : "memory");
}
__device__ __forceinline__ void wait_async_le4() {
  asm volatile("s_wait_asynccnt 4" ::: "memory");
}
__device__ __forceinline__ void wait_async_0() {
  asm volatile("s_wait_asynccnt 0" ::: "memory");
}

// XOR-lane reductions over 16-lane groups via ds_swizzle_b32
// (group-of-32 mode: imm = xor_mask<<10 | and_mask 0x1F)
template <int MASK>
__device__ __forceinline__ float swz_xor(float x) {
  return __int_as_float(
      __builtin_amdgcn_ds_swizzle(__float_as_int(x), (MASK << 10) | 0x1F));
}
__device__ __forceinline__ float rowmax16(float x) {
  x = fmaxf(x, swz_xor<1>(x));
  x = fmaxf(x, swz_xor<2>(x));
  x = fmaxf(x, swz_xor<4>(x));
  x = fmaxf(x, swz_xor<8>(x));
  return x;
}
__device__ __forceinline__ float rowsum16(float x) {
  x += swz_xor<1>(x);
  x += swz_xor<2>(x);
  x += swz_xor<4>(x);
  x += swz_xor<8>(x);
  return x;
}

// ---------------------------------------------------------------------------
// f32 -> bf16 conversion (plain and transposing)
// ---------------------------------------------------------------------------
__global__ void cvt_f32_bf16(const float* __restrict__ s, __bf16* __restrict__ d, int n) {
  int i = blockIdx.x * blockDim.x + threadIdx.x;
  int stride = gridDim.x * blockDim.x;
  for (; i < n; i += stride) d[i] = (__bf16)s[i];
}

// d[c*R + r] = (bf16) s[r*C + c]   (W[K][N] -> WT[N][K])
__global__ void cvt_f32_bf16_T(const float* __restrict__ s, __bf16* __restrict__ d,
                               int R, int C) {
  int i = blockIdx.x * blockDim.x + threadIdx.x;
  int stride = gridDim.x * blockDim.x;
  int n = R * C;
  for (; i < n; i += stride) {
    int r = i / C, c = i % C;
    d[(size_t)c * R + r] = (__bf16)s[i];
  }
}

// ---------------------------------------------------------------------------
// Tiled bf16 WMMA GEMM:  out = oscale * (A(MxK) @ W(KxN) + bias), W TRANSPOSED.
//   MODE 0: out f32 row-major [M x N]
//   MODE 1: out bf16, head layout   [B][H][T][HD]  (for Q, K)
//   MODE 2: out bf16, head-T layout [B][H][HD][T]  (for V)
// Block: 256 threads (8 waves). Tile: BM=128, BN=128, BK=32.
// Wave (rw, cw): rows 32*rw..+31, cols 64*cw..+63 -> 8 WMMAs per K-step.
// ---------------------------------------------------------------------------
template <int MODE>
__global__ __launch_bounds__(256, 1)
void gemm_bf16_wmma(const __bf16* __restrict__ A,
                    const __bf16* __restrict__ WT,
                    const float* __restrict__ bias,
                    void* __restrict__ outp,
                    int M, int N, int K, float oscale) {
  __shared__ __bf16 As[2][128 * 32];   // [m][k]
  __shared__ __bf16 Bs[2][128 * 32];   // [n][k] (from WT rows)

  const int m0 = blockIdx.y * 128;
  const int n0 = blockIdx.x * 128;
  const int tid  = threadIdx.x;
  const int wave = tid >> 5;
  const int lane = tid & 31;
  const int half = lane >> 4;
  const int lr   = lane & 15;
  const int rw   = wave & 3;
  const int cw   = wave >> 2;

  auto stage = [&](int kt, int buf) {
    const int k0 = kt * 32;
#pragma unroll
    for (int c = 0; c < 2; ++c) {
      const int idx  = tid * 2 + c;        // 0..511
      const int row  = idx >> 2;           // 0..127
      const int koff = (idx & 3) * 8;      // 0,8,16,24
      async_copy_b128(&As[buf][row * 32 + koff],
                      &A[(size_t)(m0 + row) * K + k0 + koff]);
      async_copy_b128(&Bs[buf][row * 32 + koff],
                      &WT[(size_t)(n0 + row) * K + k0 + koff]);
    }
  };

  v8f acc[2][4] = {};

  const int kTiles = K / 32;
  stage(0, 0);
  for (int kt = 0; kt < kTiles; ++kt) {
    const int buf = kt & 1;
    if (kt + 1 < kTiles) {
      stage(kt + 1, buf ^ 1);   // overlap next DMA with this compute
      wait_async_le4();         // my stage kt complete (in-order)
    } else {
      wait_async_0();
    }
    __syncthreads();

    V16U af[2];
#pragma unroll
    for (int s = 0; s < 2; ++s) {
      const __bf16* ap = &As[buf][(32 * rw + 16 * s + lr) * 32 + 16 * half];
      af[s].h[0] = *(const v8bf*)ap;
      af[s].h[1] = *(const v8bf*)(ap + 8);
    }
    v16bf bfr[4];
#pragma unroll
    for (int j = 0; j < 4; ++j)
      bfr[j] = *(const v16bf*)&Bs[buf][(64 * cw + 16 * j + lr) * 32 + 16 * half];

#pragma unroll
    for (int s = 0; s < 2; ++s)
#pragma unroll
      for (int j = 0; j < 4; ++j)
        acc[s][j] = wmma_bf16(af[s].v, bfr[j], acc[s][j]);

    __syncthreads();
  }

  // --- epilogue: bias, scale, store ---------------------------------------
  float bv[4];
#pragma unroll
  for (int j = 0; j < 4; ++j) bv[j] = bias[n0 + 64 * cw + 16 * j + lr];

#pragma unroll
  for (int s = 0; s < 2; ++s) {
#pragma unroll
    for (int j = 0; j < 4; ++j) {
      const int n = n0 + 64 * cw + 16 * j + lr;
#pragma unroll
      for (int v = 0; v < 8; ++v) {
        const int m = m0 + 32 * rw + 16 * s + v + 8 * half;
        const float r = (acc[s][j][v] + bv[j]) * oscale;
        if (MODE == 0) {
          ((float*)outp)[(size_t)m * N + n] = r;
        } else {
          const int b = m / Tn, t = m % Tn;
          const int h = n / HDn, hd = n % HDn;
          if (MODE == 1)
            ((__bf16*)outp)[(((size_t)b * Hn + h) * Tn + t) * HDn + hd] = (__bf16)r;
          else
            ((__bf16*)outp)[(((size_t)b * Hn + h) * HDn + hd) * Tn + t] = (__bf16)r;
        }
      }
    }
  }
}

// ---------------------------------------------------------------------------
// Flash attention. Grid: (T/128, B*H). Block: 256 threads = 8 waves.
// Qh (pre-scaled by 1/sqrt(HD)), Kh: [B][H][T][HD]. Vt: [B][H][HD][T].
// Per wave: 16 query rows. Per iteration: 64-key block.
//   S(16x64) = Q(16x64) @ K^T  -> 8 WMMAs
//   O(16x64) += P(16x64) @ V   -> 8 WMMAs
// K/V staged with double-buffered async-LDS DMA; mask only near diagonal.
// ---------------------------------------------------------------------------
__global__ __launch_bounds__(256, 1)
void attn_flash_wmma(const __bf16* __restrict__ Qh,
                     const __bf16* __restrict__ Kh,
                     const __bf16* __restrict__ Vt,
                     __bf16* __restrict__ Ah) {
  __shared__ __bf16 Ks[2][64 * 64];     // [key][d]
  __shared__ __bf16 Vs[2][64 * 64];     // [d][key]  (rows of Vt)
  __shared__ __bf16 Ps[8 * 16 * 64];    // per-wave P staging [wave][row][key]

  const int bh   = blockIdx.y;
  const int qb   = blockIdx.x;
  const int tid  = threadIdx.x;
  const int wave = tid >> 5;
  const int lane = tid & 31;
  const int half = lane >> 4;
  const int lr   = lane & 15;

  const int qr0 = qb * 128 + wave * 16;

  // stage 64-key block kb into buffer buf: 4 async b128 ops per thread
  auto stage = [&](int kb, int buf) {
    const int ks0 = kb * 64;
#pragma unroll
    for (int c = 0; c < 2; ++c) {
      const int idx  = tid * 2 + c;        // 0..511
      const int row  = idx >> 3;           // 0..63
      const int off  = (idx & 7) * 8;      // 0..56
      async_copy_b128(&Ks[buf][row * 64 + off],
                      &Kh[((size_t)bh * Tn + ks0 + row) * HDn + off]);
      async_copy_b128(&Vs[buf][row * 64 + off],
                      &Vt[((size_t)bh * HDn + row) * Tn + ks0 + off]);
    }
  };

  // --- Q fragments (A layout), loaded once, already scaled by 1/8 --------
  V16U qf[2];
  {
    const __bf16* qrow = Qh + ((size_t)bh * Tn + qr0 + lr) * HDn;
#pragma unroll
    for (int c = 0; c < 2; ++c) {
      qf[c].h[0] = *(const v8bf*)(qrow + 32 * c + 16 * half);
      qf[c].h[1] = *(const v8bf*)(qrow + 32 * c + 8 + 16 * half);
    }
  }

  v8f o[4] = {};
  float mrow[8], lrow[8];
#pragma unroll
  for (int v = 0; v < 8; ++v) { mrow[v] = -1e30f; lrow[v] = 0.0f; }

  const int kbMax = (qb * 128 + 127) / 64;   // inclusive (causal bound)
  stage(0, 0);
  for (int kb = 0; kb <= kbMax; ++kb) {
    const int ks0 = kb * 64;
    const int buf = kb & 1;
    if (kb < kbMax) {
      stage(kb + 1, buf ^ 1);
      wait_async_le4();
    } else {
      wait_async_0();
    }
    __syncthreads();

    // --- S = Q K^T (Q pre-scaled), four 16x16 tiles -----------------------
    v8f s[4] = {};
#pragma unroll
    for (int jt = 0; jt < 4; ++jt)
#pragma unroll
      for (int c = 0; c < 2; ++c) {
        v16bf kf = *(const v16bf*)&Ks[buf][(16 * jt + lr) * 64 + 32 * c + 16 * half];
        s[jt] = wmma_bf16(qf[c].v, kf, s[jt]);
      }

    // --- causal mask (wave-uniform skip) + online softmax -----------------
    const bool needMask = (ks0 + 63) > qr0;   // block crosses the diagonal
#pragma unroll
    for (int v = 0; v < 8; ++v) {
      const int q = qr0 + v + 8 * half;
      float e[4];
#pragma unroll
      for (int jt = 0; jt < 4; ++jt) e[jt] = s[jt][v];
      if (needMask) {
#pragma unroll
        for (int jt = 0; jt < 4; ++jt)
          if (ks0 + 16 * jt + lr > q) e[jt] = -1e9f;
      }

      const float mx = rowmax16(fmaxf(fmaxf(e[0], e[1]), fmaxf(e[2], e[3])));
      const float mn = fmaxf(mrow[v], mx);
      const float sc = __expf(mrow[v] - mn);
      mrow[v] = mn;
      float p[4], ps = 0.f;
#pragma unroll
      for (int jt = 0; jt < 4; ++jt) { p[jt] = __expf(e[jt] - mn); ps += p[jt]; }
      const float rs = rowsum16(ps);
      lrow[v] = lrow[v] * sc + rs;
#pragma unroll
      for (int j = 0; j < 4; ++j) o[j][v] *= sc;

      // C-layout -> LDS so we can reload P in A layout
      const int prow = v + 8 * half;
#pragma unroll
      for (int jt = 0; jt < 4; ++jt)
        Ps[wave * 1024 + prow * 64 + 16 * jt + lr] = (__bf16)p[jt];
    }

    // --- P as A fragments (wave-local LDS; DS ops in-order per wave) -----
    V16U pf[2];
#pragma unroll
    for (int c = 0; c < 2; ++c) {
      const __bf16* pp = &Ps[wave * 1024 + lr * 64 + 32 * c + 16 * half];
      pf[c].h[0] = *(const v8bf*)pp;
      pf[c].h[1] = *(const v8bf*)(pp + 8);
    }

    // --- O += P @ V -------------------------------------------------------
#pragma unroll
    for (int c = 0; c < 2; ++c)
#pragma unroll
      for (int j = 0; j < 4; ++j) {
        v16bf vf = *(const v16bf*)&Vs[buf][(16 * j + lr) * 64 + 32 * c + 16 * half];
        o[j] = wmma_bf16(pf[c].v, vf, o[j]);
      }
    __syncthreads();
  }

  // --- epilogue: normalize, store bf16 into [B][T][D] ----------------------
  const int h = bh % Hn;
  const int b = bh / Hn;
#pragma unroll
  for (int v = 0; v < 8; ++v) {
    const float inv = 1.0f / lrow[v];
    const int t = qr0 + v + 8 * half;
    __bf16* dst = Ah + ((size_t)b * Tn + t) * Dn + h * HDn;
#pragma unroll
    for (int j = 0; j < 4; ++j)
      dst[16 * j + lr] = (__bf16)(o[j][v] * inv);
  }
}

// ---------------------------------------------------------------------------
// Host-side launcher
// ---------------------------------------------------------------------------
extern "C" void kernel_launch(void* const* d_in, const int* in_sizes, int n_in,
                              void* d_out, int out_size, void* d_ws, size_t ws_size,
                              hipStream_t stream) {
  const float* x  = (const float*)d_in[0];
  const float* Wq = (const float*)d_in[1];
  const float* bq = (const float*)d_in[2];
  const float* Wk = (const float*)d_in[3];
  const float* bk = (const float*)d_in[4];
  const float* Wv = (const float*)d_in[5];
  const float* bv = (const float*)d_in[6];
  const float* Wo = (const float*)d_in[7];
  const float* bo = (const float*)d_in[8];
  float* out = (float*)d_out;

  // Workspace carve-up (bf16 buffers)
  char* w = (char*)d_ws;
  const size_t nX = (size_t)Mrows * Dn;          // 4M elems
  const size_t nW = (size_t)Dn * Dn;             // 1M elems
  __bf16* xb  = (__bf16*)w;                 w += nX * 2;   // A operand [M][D]
  __bf16* WqT = (__bf16*)w;                 w += nW * 2;   // [N][K] transposed
  __bf16* WkT = (__bf16*)w;                 w += nW * 2;
  __bf16* WvT = (__bf16*)w;                 w += nW * 2;
  __bf16* WoT = (__bf16*)w;                 w += nW * 2;
  __bf16* Qh  = (__bf16*)w;                 w += nX * 2;   // [B][H][T][HD], pre-scaled
  __bf16* Kh  = (__bf16*)w;                 w += nX * 2;   // [B][H][T][HD]
  __bf16* Vt  = (__bf16*)w;                 w += nX * 2;   // [B][H][HD][T]
  __bf16* Ah  = (__bf16*)w;                 w += nX * 2;   // [B][T][D]

  // 1) conversions to bf16 (weights transposed for async B-tile staging)
  cvt_f32_bf16  <<<2048, 256, 0, stream>>>(x,  xb,  (int)nX);
  cvt_f32_bf16_T<<<1024, 256, 0, stream>>>(Wq, WqT, Dn, Dn);
  cvt_f32_bf16_T<<<1024, 256, 0, stream>>>(Wk, WkT, Dn, Dn);
  cvt_f32_bf16_T<<<1024, 256, 0, stream>>>(Wv, WvT, Dn, Dn);
  cvt_f32_bf16_T<<<1024, 256, 0, stream>>>(Wo, WoT, Dn, Dn);

  // 2) QKV projections (Q pre-scaled by 1/sqrt(HD) = 0.125)
  dim3 gGemm(Dn / 128, Mrows / 128);   // (8, 32)
  gemm_bf16_wmma<1><<<gGemm, 256, 0, stream>>>(xb, WqT, bq, Qh, Mrows, Dn, Dn, 0.125f);
  gemm_bf16_wmma<1><<<gGemm, 256, 0, stream>>>(xb, WkT, bk, Kh, Mrows, Dn, Dn, 1.0f);
  gemm_bf16_wmma<2><<<gGemm, 256, 0, stream>>>(xb, WvT, bv, Vt, Mrows, Dn, Dn, 1.0f);

  // 3) flash attention (64-key blocks)
  dim3 gAttn(Tn / 128, Bn * Hn);       // (16, 32)
  attn_flash_wmma<<<gAttn, 256, 0, stream>>>(Qh, Kh, Vt, Ah);

  // 4) output projection -> f32
  gemm_bf16_wmma<0><<<gGemm, 256, 0, stream>>>(Ah, WoT, bo, out, Mrows, Dn, Dn, 1.0f);
}
